// EdgeLLMHybridBlock_69715909149350
// MI455X (gfx1250) — matrix operand order repaired and checked by reading, not compiled, not run on hardware
//
#include <hip/hip_runtime.h>
#include <cstdint>

typedef __attribute__((ext_vector_type(16))) __bf16 bf16x16;
typedef __attribute__((ext_vector_type(8)))  float  v8f;
typedef __bf16 bf16_t;

#define B_    2
#define S_    2048
#define H_    2048
#define HQ_   16
#define HKV_  4
#define D_    128
#define MAXP_ 4096
#define M_    (B_ * S_)                       // 4096 rows
#define NQKV_ (HQ_ * D_ + 2 * HKV_ * D_)      // 3072
#define ATTN_ELEMS  ((size_t)B_ * S_ * H_)                    // 8388608
#define CACHE_ELEMS ((size_t)B_ * 2 * HKV_ * MAXP_ * D_)      // 8388608
#define QK_SCALE 0.08838834764831845f          // 1/sqrt(128)

// ---------------- helpers ----------------

__device__ inline bf16_t f2bf(float x) {
  union { float f; unsigned int u; } v; v.f = x;
  unsigned int r = v.u + 0x7FFFu + ((v.u >> 16) & 1u);   // round-to-nearest-even
  union { unsigned short s; bf16_t b; } o; o.s = (unsigned short)(r >> 16);
  return o.b;
}

// A fragment: 16x32 bf16, row-major source, ldA elements (contiguous k order: k = 16*(lane/16) + s)
__device__ inline bf16x16 load_a_frag(const bf16_t* __restrict__ A, int ldA,
                                      int m0, int k0, int lane) {
  union { bf16x16 v; unsigned int u[8]; } f;
  int m  = m0 + (lane & 15);
  int kh = (lane >> 4) * 16;
  const bf16_t* base = A + (size_t)m * ldA + k0;
#pragma unroll
  for (int j = 0; j < 4; ++j) {
    f.u[j]     = *(const unsigned int*)(base + kh + 2 * j);
    f.u[4 + j] = *(const unsigned int*)(base + 8 + kh + 2 * j);
  }
  return f.v;
}

// B fragment for C = A * W^T : WMMA-B[k][n] = W[n][k], W row-major NxK, ldW elems
__device__ inline bf16x16 load_bT_frag(const bf16_t* __restrict__ W, int ldW,
                                       int n0, int k0, int lane) {
  union { bf16x16 v; unsigned int u[8]; } f;
  int n  = n0 + (lane & 15);
  int kb = k0 + (lane >> 4) * 16;
  const bf16_t* base = W + (size_t)n * ldW + kb;
#pragma unroll
  for (int j = 0; j < 8; ++j) f.u[j] = *(const unsigned int*)(base + 2 * j);
  return f.v;
}

// B fragment from a 64x32 LDS-staged tile
__device__ inline bf16x16 lds_b_frag(const bf16_t* Bs, int nloc, int lane) {
  union { bf16x16 v; unsigned int u[8]; } f;
  int n  = nloc + (lane & 15);
  int kb = (lane >> 4) * 16;
  const bf16_t* base = Bs + n * 32 + kb;
#pragma unroll
  for (int j = 0; j < 8; ++j) f.u[j] = *(const unsigned int*)(base + 2 * j);
  return f.v;
}

__device__ inline bf16x16 load_p_frag(const bf16_t* __restrict__ p /*16x32*/, int lane) {
  union { bf16x16 v; unsigned int u[8]; } f;
  int m  = lane & 15;
  int kh = (lane >> 4) * 16;
  const bf16_t* row = p + m * 32;
#pragma unroll
  for (int j = 0; j < 4; ++j) {
    f.u[j]     = *(const unsigned int*)(row + kh + 2 * j);
    f.u[4 + j] = *(const unsigned int*)(row + 8 + kh + 2 * j);
  }
  return f.v;
}

__device__ inline float half_max16(float x) {   // reduce over 16-lane half
#pragma unroll
  for (int off = 8; off; off >>= 1) x = fmaxf(x, __shfl_xor(x, off, 32));
  return x;
}
__device__ inline float half_sum16(float x) {
#pragma unroll
  for (int off = 8; off; off >>= 1) x += __shfl_xor(x, off, 32);
  return x;
}

// ---------------- utility kernels ----------------

__global__ void cvt_f32_bf16(const float* __restrict__ in, bf16_t* __restrict__ out, size_t n) {
  size_t i = (size_t)blockIdx.x * blockDim.x + threadIdx.x;
  size_t st = (size_t)gridDim.x * blockDim.x;
  for (; i < n; i += st) out[i] = f2bf(in[i]);
}

__global__ void copy_f32(const float* __restrict__ in, float* __restrict__ out, size_t n4) {
  size_t i = (size_t)blockIdx.x * blockDim.x + threadIdx.x;
  size_t st = (size_t)gridDim.x * blockDim.x;
  const float4* s = (const float4*)in;
  float4* d = (float4*)out;
  for (; i < n4; i += st) d[i] = s[i];
}

// ---------------- block-cooperative GEMM ----------------
// C(MxN) f32 = A(MxK) bf16 * W(NxK)^T bf16.
// Block = 256 thr (8 waves); block tile 128x64 (wave: 16x64, 4 accumulators).
// B tile (64 rows x 32 k = 4KB) is DMA'd to LDS with global_load_async_to_lds_b128,
// double-buffered (ASYNCcnt), so all 8 waves share one copy of the weights.
// Requires M%128==0, N%64==0, K%32==0, grid == (M/128)*(N/64).

__global__ __launch_bounds__(256)
void gemm_bf16_nt(const bf16_t* __restrict__ A, const bf16_t* __restrict__ W,
                  float* __restrict__ C, int M, int N, int K) {
  __shared__ bf16_t Bs[2][64][32];                 // 8 KB double buffer
  int tid  = threadIdx.x;
  int wid  = tid >> 5, lane = tid & 31;
  int bn   = N >> 6;
  int m0   = (blockIdx.x / bn) * 128 + wid * 16;
  int n0   = (blockIdx.x % bn) << 6;

  int brow = tid >> 2;                             // 0..63: B tile row
  int bseg = (tid & 3) * 8;                        // 8-elem (16B) segment in row
  unsigned lds0 = (unsigned)(uintptr_t)(&Bs[0][0][0]) + (unsigned)tid * 16u;
  unsigned lds1 = (unsigned)(uintptr_t)(&Bs[1][0][0]) + (unsigned)tid * 16u;
  const bf16_t* gB = W + (size_t)(n0 + brow) * K + bseg;

  // prologue: stage k0 = 0 into buffer 0
  {
    unsigned long long g = (unsigned long long)(uintptr_t)gB;
    asm volatile("global_load_async_to_lds_b128 %0, %1, off"
                 :: "v"(lds0), "v"(g) : "memory");
  }

  v8f acc0 = {}, acc1 = {}, acc2 = {}, acc3 = {};
  for (int k0 = 0; k0 < K; k0 += 32) {
    int cur = (k0 >> 5) & 1;
    asm volatile("s_wait_asynccnt 0" ::: "memory");
    __syncthreads();                               // staged tile visible to all waves
    if (k0 + 32 < K) {                             // stream next tile into other buffer
      unsigned long long g = (unsigned long long)(uintptr_t)(gB + k0 + 32);
      asm volatile("global_load_async_to_lds_b128 %0, %1, off"
                   :: "v"(cur ? lds0 : lds1), "v"(g) : "memory");
    }
    if (k0 + 64 <= K)
      __builtin_prefetch(A + (size_t)(m0 + (lane & 15)) * K + k0 + 32, 0, 3);

    bf16x16 a = load_a_frag(A, K, m0, k0, lane);
    const bf16_t* bs = &Bs[cur][0][0];
    bf16x16 b0 = lds_b_frag(bs, 0,  lane);
    bf16x16 b1 = lds_b_frag(bs, 16, lane);
    bf16x16 b2 = lds_b_frag(bs, 32, lane);
    bf16x16 b3 = lds_b_frag(bs, 48, lane);
    acc0 = __builtin_amdgcn_wmma_f32_16x16x32_bf16(false, a, false, b0, (short)0, acc0, false, false);
    acc1 = __builtin_amdgcn_wmma_f32_16x16x32_bf16(false, a, false, b1, (short)0, acc1, false, false);
    acc2 = __builtin_amdgcn_wmma_f32_16x16x32_bf16(false, a, false, b2, (short)0, acc2, false, false);
    acc3 = __builtin_amdgcn_wmma_f32_16x16x32_bf16(false, a, false, b3, (short)0, acc3, false, false);
    __syncthreads();                               // reads done before buffer reuse
  }

  int n = lane & 15, mo = (lane >> 4) * 8;
  v8f accs[4] = {acc0, acc1, acc2, acc3};
#pragma unroll
  for (int t = 0; t < 4; ++t)
#pragma unroll
    for (int r = 0; r < 8; ++r)
      C[(size_t)(m0 + mo + r) * N + n0 + 16 * t + n] = accs[t][r];
}

// ---------------- RMSNorm + RoPE + cache write ----------------
// wave per (row, head): heads 0..15 = Q, 16..19 = K, 20..23 = V.
// V is additionally written TRANSPOSED ([b][hkv][D][S]) so the flash kernel's
// P*V B-fragments are contiguous loads.

__global__ void rmsnorm_rope_kernel(const float* __restrict__ qkv,
                                    const float* __restrict__ cs,
                                    const float* __restrict__ qw,
                                    const float* __restrict__ kw,
                                    const int* __restrict__ start_idx,
                                    bf16_t* __restrict__ qout,
                                    bf16_t* __restrict__ kout,
                                    bf16_t* __restrict__ vt,
                                    float* __restrict__ cache) {
  int wave = (int)((blockIdx.x * blockDim.x + threadIdx.x) >> 5);
  int lane = threadIdx.x & 31;
  if (wave >= M_ * 24) return;
  int row = wave / 24;
  int h   = wave % 24;
  int b = row >> 11, s = row & (S_ - 1);
  int d = lane * 2;                                   // 0..62 (first half)
  const float* rowp = qkv + (size_t)row * NQKV_;
  int kind, hd, col;
  if (h < 16)      { kind = 0; hd = h;      col = hd * D_; }
  else if (h < 20) { kind = 1; hd = h - 16; col = HQ_ * D_ + hd * D_; }
  else             { kind = 2; hd = h - 20; col = HQ_ * D_ + HKV_ * D_ + hd * D_; }

  float x0 = rowp[col + d],      x1 = rowp[col + d + 1];
  float y0 = rowp[col + 64 + d], y1 = rowp[col + 64 + d + 1];

  if (kind < 2) {                                     // RMSNorm + RoPE for q/k
    float ss = x0 * x0 + x1 * x1 + y0 * y0 + y1 * y1;
#pragma unroll
    for (int off = 16; off; off >>= 1) ss += __shfl_xor(ss, off, 32);
    float rinv = rsqrtf(ss * (1.0f / 128.0f) + 1e-6f);
    const float* w = (kind == 0) ? qw : kw;
    x0 *= rinv * w[d];      x1 *= rinv * w[d + 1];
    y0 *= rinv * w[64 + d]; y1 *= rinv * w[64 + d + 1];
    const float* c = cs + (size_t)row * D_;
    float c0 = c[d], c1 = c[d + 1], s0 = c[64 + d], s1 = c[64 + d + 1];
    float o0 = x0 * c0 - y0 * s0, o1 = x1 * c1 - y1 * s1;
    float p0 = y0 * c0 + x0 * s0, p1 = y1 * c1 + x1 * s1;
    x0 = o0; x1 = o1; y0 = p0; y1 = p1;
  }

  if (kind == 0) {
    bf16_t* q = qout + (size_t)row * (HQ_ * D_) + hd * D_;
    q[d] = f2bf(x0); q[d + 1] = f2bf(x1); q[64 + d] = f2bf(y0); q[64 + d + 1] = f2bf(y1);
  } else if (kind == 1) {
    bf16_t* o = kout + ((size_t)(b * HKV_ + hd) * S_ + s) * D_;
    o[d] = f2bf(x0); o[d + 1] = f2bf(x1); o[64 + d] = f2bf(y0); o[64 + d + 1] = f2bf(y1);
    int pos = start_idx[b] + s;
    float* cc = cache + (((size_t)(b * 2 + 0) * HKV_ + hd) * MAXP_ + pos) * D_;
    cc[d] = x0; cc[d + 1] = x1; cc[64 + d] = y0; cc[64 + d + 1] = y1;
  } else {                                            // V: transposed bf16 + cache
    bf16_t* o = vt + (size_t)(b * HKV_ + hd) * D_ * S_;
    o[(size_t)d * S_ + s]        = f2bf(x0);
    o[(size_t)(d + 1) * S_ + s]  = f2bf(x1);
    o[(size_t)(64 + d) * S_ + s]     = f2bf(y0);
    o[(size_t)(64 + d + 1) * S_ + s] = f2bf(y1);
    int pos = start_idx[b] + s;
    float* cc = cache + (((size_t)(b * 2 + 1) * HKV_ + hd) * MAXP_ + pos) * D_;
    cc[d] = x0; cc[d + 1] = x1; cc[64 + d] = y0; cc[64 + d + 1] = y1;
  }
}

// ---------------- Flash attention (wave per 16 query rows) ----------------

__global__ __launch_bounds__(128)
void flash_attn_kernel(const bf16_t* __restrict__ Q, const bf16_t* __restrict__ Kb,
                       const bf16_t* __restrict__ Vt, const int* __restrict__ ctx_len,
                       float* __restrict__ Out) {
  __shared__ bf16_t plds[4][16][32];
  int wid  = threadIdx.x >> 5;
  int lane = threadIdx.x & 31;
  int gw = blockIdx.x * 4 + wid;                 // 0..4095
  if (gw >= B_ * HQ_ * (S_ / 16)) return;
  int qt  = gw & 127;
  int bh  = gw >> 7;
  int b   = bh >> 4, hq = bh & 15;
  int hkv = hq >> 2;
  int q0  = qt * 16;

  const bf16_t* Qh = Q + hq * D_;                              // row-major M x 2048
  const bf16_t* Kh = Kb + (size_t)(b * HKV_ + hkv) * S_ * D_;  // S x 128
  const bf16_t* Vh = Vt + (size_t)(b * HKV_ + hkv) * D_ * S_;  // transposed: 128 x S

  bf16x16 qf0 = load_a_frag(Qh, HQ_ * D_, b * S_ + q0, 0,  lane);
  bf16x16 qf1 = load_a_frag(Qh, HQ_ * D_, b * S_ + q0, 32, lane);
  bf16x16 qf2 = load_a_frag(Qh, HQ_ * D_, b * S_ + q0, 64, lane);
  bf16x16 qf3 = load_a_frag(Qh, HQ_ * D_, b * S_ + q0, 96, lane);

  v8f acc[8]; float m_r[8], l_r[8];
#pragma unroll
  for (int t = 0; t < 8; ++t) acc[t] = {};
#pragma unroll
  for (int r = 0; r < 8; ++r) { m_r[r] = -1e30f; l_r[r] = 0.0f; }

  int ctx  = ctx_len[b];
  int klo  = lane & 15, mo = (lane >> 4) * 8;
  int nkb  = (q0 + 15 + 32) >> 5;                // blocks with key0 <= q0+15

  for (int kb = 0; kb < nkb; ++kb) {
    int key0 = kb * 32;
    v8f sc0 = {}, sc1 = {};
#pragma unroll
    for (int j = 0; j < 4; ++j) {
      bf16x16 qf = (j == 0) ? qf0 : (j == 1) ? qf1 : (j == 2) ? qf2 : qf3;
      bf16x16 k0f = load_bT_frag(Kh, D_, key0,      32 * j, lane);
      sc0 = __builtin_amdgcn_wmma_f32_16x16x32_bf16(false, qf, false, k0f, (short)0, sc0, false, false);
      bf16x16 k1f = load_bT_frag(Kh, D_, key0 + 16, 32 * j, lane);
      sc1 = __builtin_amdgcn_wmma_f32_16x16x32_bf16(false, qf, false, k1f, (short)0, sc1, false, false);
    }
    int nk0 = key0 + klo, nk1 = nk0 + 16;
#pragma unroll
    for (int r = 0; r < 8; ++r) {
      int qi = q0 + mo + r;
      float s0 = sc0[r] * QK_SCALE, s1 = sc1[r] * QK_SCALE;
      if (nk0 > qi || nk0 >= ctx) s0 = -1e30f;
      if (nk1 > qi || nk1 >= ctx) s1 = -1e30f;
      float mx = half_max16(fmaxf(s0, s1));
      float mn = fmaxf(m_r[r], mx);
      float cf = __expf(m_r[r] - mn);
      float p0 = __expf(s0 - mn), p1 = __expf(s1 - mn);
      l_r[r] = l_r[r] * cf + half_sum16(p0 + p1);
      m_r[r] = mn;
#pragma unroll
      for (int t = 0; t < 8; ++t) acc[t][r] *= cf;
      plds[wid][mo + r][klo]      = f2bf(p0);
      plds[wid][mo + r][klo + 16] = f2bf(p1);
    }
    // wave-private LDS tile: enforce DS-store -> DS-load ordering (CDNA5 split counter)
    asm volatile("s_wait_dscnt 0" ::: "memory");
    bf16x16 pf = load_p_frag(&plds[wid][0][0], lane);
#pragma unroll
    for (int t = 0; t < 8; ++t) {
      // B[k][n] = V[key0+k][16t+n] = Vt[(16t+n)*S + key0+k] -> contiguous in k
      bf16x16 vf = load_bT_frag(Vh, S_, 16 * t, key0, lane);
      acc[t] = __builtin_amdgcn_wmma_f32_16x16x32_bf16(false, pf, false, vf, (short)0, acc[t], false, false);
    }
  }

#pragma unroll
  for (int t = 0; t < 8; ++t)
#pragma unroll
    for (int r = 0; r < 8; ++r)
      Out[(size_t)(b * S_ + q0 + mo + r) * (HQ_ * D_) + hq * D_ + 16 * t + klo] =
          acc[t][r] / l_r[r];
}

// ---------------- launcher ----------------

extern "C" void kernel_launch(void* const* d_in, const int* in_sizes, int n_in,
                              void* d_out, int out_size, void* d_ws, size_t ws_size,
                              hipStream_t stream) {
  const float* hs    = (const float*)d_in[0];
  const float* past  = (const float*)d_in[1];
  const float* cs    = (const float*)d_in[2];
  const float* Wq    = (const float*)d_in[3];
  const float* Wk    = (const float*)d_in[4];
  const float* Wv    = (const float*)d_in[5];
  const float* Wo    = (const float*)d_in[6];
  const float* qnw   = (const float*)d_in[7];
  const float* knw   = (const float*)d_in[8];
  const int*   ctx   = (const int*)d_in[9];
  const int*   start = (const int*)d_in[10];

  float* out_attn  = (float*)d_out;
  float* out_cache = out_attn + ATTN_ELEMS;

  char* ws = (char*)d_ws;
  bf16_t* hs_bf   = (bf16_t*)(ws + 0);            // 16.78 MB
  bf16_t* wqkv    = (bf16_t*)(ws + 16777216);     // 12.58 MB (stacked Wq|Wk|Wv)
  bf16_t* wo_bf   = (bf16_t*)(ws + 29360128);     //  8.39 MB
  float*  qkv     = (float*) (ws + 37748736);     // 50.33 MB
  bf16_t* q_bf    = (bf16_t*)(ws + 88080384);     // 16.78 MB
  bf16_t* k_bf    = (bf16_t*)(ws + 104857600);    //  4.19 MB
  bf16_t* vt_bf   = (bf16_t*)(ws + 109051904);    //  4.19 MB  (total 108 MiB)
  float*  attn    = qkv;                          // reuse: qkv dead after rope
  bf16_t* attn_bf = hs_bf;                        // reuse: hs dead after gemm1

  // 1. seed present_key_value with past cache
  copy_f32<<<4096, 256, 0, stream>>>(past, out_cache, CACHE_ELEMS / 4);
  // 2. bf16 conversions
  cvt_f32_bf16<<<2048, 256, 0, stream>>>(hs, hs_bf, ATTN_ELEMS);
  cvt_f32_bf16<<<1024, 256, 0, stream>>>(Wq, wqkv, (size_t)2048 * 2048);
  cvt_f32_bf16<<<256,  256, 0, stream>>>(Wk, wqkv + (size_t)2048 * 2048, (size_t)512 * 2048);
  cvt_f32_bf16<<<256,  256, 0, stream>>>(Wv, wqkv + (size_t)2560 * 2048, (size_t)512 * 2048);
  cvt_f32_bf16<<<1024, 256, 0, stream>>>(Wo, wo_bf, (size_t)2048 * 2048);
  // 3. fused QKV projection: (4096x2048) x (3072x2048)^T, grid = (M/128)*(N/64)
  gemm_bf16_nt<<<1536, 256, 0, stream>>>(hs_bf, wqkv, qkv, M_, NQKV_, H_);
  // 4. RMSNorm + RoPE + cache update (+ V transpose)
  rmsnorm_rope_kernel<<<12288, 256, 0, stream>>>(qkv, cs, qnw, knw, start,
                                                 q_bf, k_bf, vt_bf, out_cache);
  // 5. causal GQA flash attention
  flash_attn_kernel<<<1024, 128, 0, stream>>>(q_bf, k_bf, vt_bf, ctx, attn);
  // 6. output projection: (4096x2048) x (2048x2048)^T, grid = (M/128)*(N/64)
  cvt_f32_bf16<<<2048, 256, 0, stream>>>(attn, attn_bf, ATTN_ELEMS);
  gemm_bf16_nt<<<1024, 256, 0, stream>>>(attn_bf, wo_bf, out_attn, M_, H_, H_);
}
